// LlamaForCausalLM_23287312679540
// MI455X (gfx1250) — compile-verified, matrix operand order checked by reading, not run on hardware
//
#include <hip/hip_runtime.h>

// ---- Model dims ----
#define Hdim   2048
#define NH     32
#define NKV    4
#define HD     64
#define Idim   5632
#define QKVD   2560      // (NH + 2*NKV) * HD
#define SLEN   1024
#define Bsz    2
#define NTOK   2048      // B*S
#define EPSv   1e-5f

typedef __bf16 bf16;
typedef bf16  v16bf __attribute__((ext_vector_type(16)));
typedef bf16  v8bf  __attribute__((ext_vector_type(8)));
typedef float v8f   __attribute__((ext_vector_type(8)));

// native fptrunc -> v_cvt_*bf16_f32 on gfx1250 (RNE)
static __device__ __forceinline__ bf16 f2bf(float f) { return (bf16)f; }

// -------------------- embedding gather --------------------
__global__ __launch_bounds__(256) void embed_kernel(const int* __restrict__ ids,
                                                    const float* __restrict__ emb,
                                                    float* __restrict__ resid) {
    int t = blockIdx.x;
    int id = ids[t];
    size_t src = (size_t)id * Hdim + threadIdx.x * 8;
    size_t dst = (size_t)t  * Hdim + threadIdx.x * 8;
    float4 a = *(const float4*)(emb + src);
    float4 b = *(const float4*)(emb + src + 4);
    *(float4*)(resid + dst)     = a;
    *(float4*)(resid + dst + 4) = b;
}

// -------------------- fused (residual += h), RMSNorm --------------------
__global__ __launch_bounds__(256) void rmsnorm_kernel(float* __restrict__ resid,
                                                      const float* __restrict__ hadd,
                                                      const float* __restrict__ w,
                                                      float* __restrict__ out) {
    __shared__ float red[256];
    int t = blockIdx.x, tid = threadIdx.x;
    size_t base = (size_t)t * Hdim + tid * 8;
    float x[8];
    float ss = 0.f;
#pragma unroll
    for (int j = 0; j < 8; ++j) {
        float v = resid[base + j];
        if (hadd) v += hadd[base + j];
        x[j] = v;
        ss += v * v;
    }
    if (hadd) {
#pragma unroll
        for (int j = 0; j < 8; ++j) resid[base + j] = x[j];
    }
    red[tid] = ss;
    for (int st = 128; st > 0; st >>= 1) {
        __syncthreads();
        if (tid < st) red[tid] += red[tid + st];
    }
    __syncthreads();
    float rms = rsqrtf(red[0] * (1.0f / Hdim) + EPSv);
#pragma unroll
    for (int j = 0; j < 8; ++j)
        out[base + j] = x[j] * rms * w[tid * 8 + j];
}

// -------------------- generic bf16-WMMA GEMM: C[M,N] (=|+=) A[M,K] @ W[K,N] --------------------
// Block: 256 thr = 8 waves. Block tile: 128(M) x 64(N). K step 32.
template <int MODE>   // 0: C = r ; 1: C += r
__global__ __launch_bounds__(256) void gemm_kernel(const float* __restrict__ A,
                                                   const float* __restrict__ W,
                                                   float* __restrict__ C,
                                                   int M, int N, int K) {
    __shared__ __align__(16) bf16 Wlds[64][40];   // [n][k], 8-elem pad per row
    const int tid  = threadIdx.x;
    const int lane = tid & 31, wave = tid >> 5;
    const int m0   = blockIdx.y * 128 + wave * 16;
    const int n0   = blockIdx.x * 64;
    const int row  = m0 + (lane & 15);
    const int kgrp = (lane >> 4) * 8;
    const int kk   = tid >> 3;            // 0..31 : K row of tile
    const int nb   = (tid & 7) * 8;       // 0..56 : N base of this thread's 8 cols

    v8f acc[4] = {};
    for (int k0 = 0; k0 < K; k0 += 32) {
        const float* wp = W + (size_t)(k0 + kk) * N + (n0 + nb);
        float4 w0 = *(const float4*)wp;
        float4 w1 = *(const float4*)(wp + 4);
        if (k0 + 32 < K) __builtin_prefetch(wp + (size_t)32 * N, 0, 1);
        __syncthreads();   // previous iter's LDS reads done
        float wv[8] = {w0.x, w0.y, w0.z, w0.w, w1.x, w1.y, w1.z, w1.w};
#pragma unroll
        for (int j = 0; j < 8; ++j) Wlds[nb + j][kk] = f2bf(wv[j]);
        __syncthreads();   // tile ready

        const float* ap = A + (size_t)row * K + k0 + kgrp;
        float4 a0 = *(const float4*)ap,        a1 = *(const float4*)(ap + 4);
        float4 a2 = *(const float4*)(ap + 16), a3 = *(const float4*)(ap + 20);
        v16bf af;
        af[0]=f2bf(a0.x); af[1]=f2bf(a0.y); af[2]=f2bf(a0.z); af[3]=f2bf(a0.w);
        af[4]=f2bf(a1.x); af[5]=f2bf(a1.y); af[6]=f2bf(a1.z); af[7]=f2bf(a1.w);
        af[8]=f2bf(a2.x); af[9]=f2bf(a2.y); af[10]=f2bf(a2.z); af[11]=f2bf(a2.w);
        af[12]=f2bf(a3.x); af[13]=f2bf(a3.y); af[14]=f2bf(a3.z); af[15]=f2bf(a3.w);
#pragma unroll
        for (int t = 0; t < 4; ++t) {
            const bf16* bp = &Wlds[t * 16 + (lane & 15)][kgrp];
            v8bf blo = *(const v8bf*)bp;
            v8bf bhi = *(const v8bf*)(bp + 16);
            v16bf bfr;
#pragma unroll
            for (int j = 0; j < 8; ++j) { bfr[j] = blo[j]; bfr[8 + j] = bhi[j]; }
            acc[t] = __builtin_amdgcn_wmma_f32_16x16x32_bf16(
                false, af, false, bfr, (short)0, acc[t], false, false);
        }
    }
    const int rbase = m0 + ((lane >> 4) << 3);
#pragma unroll
    for (int t = 0; t < 4; ++t) {
        const int col = n0 + t * 16 + (lane & 15);
#pragma unroll
        for (int i = 0; i < 8; ++i) {
            size_t o = (size_t)(rbase + i) * N + col;
            if (MODE == 0) C[o] = acc[t][i];
            else           C[o] += acc[t][i];
        }
    }
}

// -------------------- fused gate/up GEMM + SiLU: act = silu(hn@Wg) * (hn@Wu) --------------------
__global__ __launch_bounds__(256) void gemm_gateup_kernel(const float* __restrict__ A,
                                                          const float* __restrict__ W,
                                                          float* __restrict__ act) {
    __shared__ __align__(16) bf16 Glds[64][40];
    __shared__ __align__(16) bf16 Ulds[64][40];
    const int N2 = 2 * Idim;
    const int tid  = threadIdx.x;
    const int lane = tid & 31, wave = tid >> 5;
    const int m0   = blockIdx.y * 128 + wave * 16;
    const int n0   = blockIdx.x * 64;
    const int row  = m0 + (lane & 15);
    const int kgrp = (lane >> 4) * 8;
    const int kk   = tid >> 3;
    const int nb   = (tid & 7) * 8;

    v8f accg[4] = {}, accu[4] = {};
    for (int k0 = 0; k0 < Hdim; k0 += 32) {
        const float* gp = W + (size_t)(k0 + kk) * N2 + (n0 + nb);
        const float* up = gp + Idim;
        float4 g0 = *(const float4*)gp, g1 = *(const float4*)(gp + 4);
        float4 u0 = *(const float4*)up, u1 = *(const float4*)(up + 4);
        if (k0 + 32 < Hdim) {
            __builtin_prefetch(gp + (size_t)32 * N2, 0, 1);
            __builtin_prefetch(up + (size_t)32 * N2, 0, 1);
        }
        __syncthreads();
        float gv[8] = {g0.x, g0.y, g0.z, g0.w, g1.x, g1.y, g1.z, g1.w};
        float uv[8] = {u0.x, u0.y, u0.z, u0.w, u1.x, u1.y, u1.z, u1.w};
#pragma unroll
        for (int j = 0; j < 8; ++j) {
            Glds[nb + j][kk] = f2bf(gv[j]);
            Ulds[nb + j][kk] = f2bf(uv[j]);
        }
        __syncthreads();

        const float* ap = A + (size_t)row * Hdim + k0 + kgrp;
        float4 a0 = *(const float4*)ap,        a1 = *(const float4*)(ap + 4);
        float4 a2 = *(const float4*)(ap + 16), a3 = *(const float4*)(ap + 20);
        v16bf af;
        af[0]=f2bf(a0.x); af[1]=f2bf(a0.y); af[2]=f2bf(a0.z); af[3]=f2bf(a0.w);
        af[4]=f2bf(a1.x); af[5]=f2bf(a1.y); af[6]=f2bf(a1.z); af[7]=f2bf(a1.w);
        af[8]=f2bf(a2.x); af[9]=f2bf(a2.y); af[10]=f2bf(a2.z); af[11]=f2bf(a2.w);
        af[12]=f2bf(a3.x); af[13]=f2bf(a3.y); af[14]=f2bf(a3.z); af[15]=f2bf(a3.w);
#pragma unroll
        for (int t = 0; t < 4; ++t) {
            const bf16* gb = &Glds[t * 16 + (lane & 15)][kgrp];
            const bf16* ub = &Ulds[t * 16 + (lane & 15)][kgrp];
            v8bf glo = *(const v8bf*)gb, ghi = *(const v8bf*)(gb + 16);
            v8bf ulo = *(const v8bf*)ub, uhi = *(const v8bf*)(ub + 16);
            v16bf gf, uf;
#pragma unroll
            for (int j = 0; j < 8; ++j) {
                gf[j] = glo[j]; gf[8 + j] = ghi[j];
                uf[j] = ulo[j]; uf[8 + j] = uhi[j];
            }
            accg[t] = __builtin_amdgcn_wmma_f32_16x16x32_bf16(
                false, af, false, gf, (short)0, accg[t], false, false);
            accu[t] = __builtin_amdgcn_wmma_f32_16x16x32_bf16(
                false, af, false, uf, (short)0, accu[t], false, false);
        }
    }
    const int rbase = m0 + ((lane >> 4) << 3);
#pragma unroll
    for (int t = 0; t < 4; ++t) {
        const int col = n0 + t * 16 + (lane & 15);
#pragma unroll
        for (int i = 0; i < 8; ++i) {
            float g = accg[t][i];
            float u = accu[t][i];
            float s = g / (1.0f + __expf(-g));    // SiLU
            act[(size_t)(rbase + i) * Idim + col] = s * u;
        }
    }
}

// -------------------- RoPE in place on q and k portions of qkv --------------------
__global__ __launch_bounds__(256) void rope_kernel(float* __restrict__ qkv,
                                                   const int* __restrict__ pos) {
    int idx = blockIdx.x * blockDim.x + threadIdx.x;
    int i  = idx & 31;
    int hh = (idx >> 5) % (NH + NKV);                  // q heads 0..31, k heads 32..35
    int t  = idx / ((NH + NKV) * 32);
    int s  = t & (SLEN - 1);
    float p = (float)pos[s];
    float invf = __powf(10000.0f, -(float)i / 32.0f);
    float ang = p * invf;
    float c = __cosf(ang), sn = __sinf(ang);
    size_t base = (size_t)t * QKVD + hh * HD;
    float x1 = qkv[base + i];
    float x2 = qkv[base + 32 + i];
    qkv[base + i]      = x1 * c - x2 * sn;
    qkv[base + 32 + i] = x2 * c + x1 * sn;
}

// -------------------- WMMA flash attention --------------------
// Block: 256 thr = 8 waves; one (b, head); 128 q rows (16 per wave).
// K chunk = 32 keys staged to LDS per block iteration.
__global__ __launch_bounds__(256) void attn_kernel(const float* __restrict__ qkv,
                                                   float* __restrict__ attn) {
    __shared__ __align__(16) bf16 Klds[32][72];      // [k][d] (pad 64->72)
    __shared__ __align__(16) bf16 Vt[64][40];        // [d][k] (pad 32->40)
    __shared__ __align__(16) bf16 Plds[8][16][40];   // per-wave P re-striping

    const int tid  = threadIdx.x;
    const int lane = tid & 31, wave = tid >> 5;
    const int head = blockIdx.y & (NH - 1);
    const int b    = blockIdx.y >> 5;
    const int kh   = head >> 3;                      // NH/NKV = 8
    const int qbase = blockIdx.x * 128 + wave * 16;
    const int l16  = lane & 15;
    const int grp  = (lane >> 4) * 8;
    const int rh   = (lane >> 4) << 3;               // row offset of C-layout half

    // ---- preload Q fragments (16 rows x 64 d, two 16x32 A-fragments) ----
    v16bf aq0, aq1;
    {
        const float* qp = qkv + (size_t)(b * SLEN + qbase + l16) * QKVD + head * HD;
        float4 x0 = *(const float4*)(qp + grp);
        float4 x1 = *(const float4*)(qp + grp + 4);
        float4 x2 = *(const float4*)(qp + grp + 16);
        float4 x3 = *(const float4*)(qp + grp + 20);
        float4 y0 = *(const float4*)(qp + 32 + grp);
        float4 y1 = *(const float4*)(qp + 32 + grp + 4);
        float4 y2 = *(const float4*)(qp + 32 + grp + 16);
        float4 y3 = *(const float4*)(qp + 32 + grp + 20);
        aq0[0]=f2bf(x0.x); aq0[1]=f2bf(x0.y); aq0[2]=f2bf(x0.z); aq0[3]=f2bf(x0.w);
        aq0[4]=f2bf(x1.x); aq0[5]=f2bf(x1.y); aq0[6]=f2bf(x1.z); aq0[7]=f2bf(x1.w);
        aq0[8]=f2bf(x2.x); aq0[9]=f2bf(x2.y); aq0[10]=f2bf(x2.z); aq0[11]=f2bf(x2.w);
        aq0[12]=f2bf(x3.x); aq0[13]=f2bf(x3.y); aq0[14]=f2bf(x3.z); aq0[15]=f2bf(x3.w);
        aq1[0]=f2bf(y0.x); aq1[1]=f2bf(y0.y); aq1[2]=f2bf(y0.z); aq1[3]=f2bf(y0.w);
        aq1[4]=f2bf(y1.x); aq1[5]=f2bf(y1.y); aq1[6]=f2bf(y1.z); aq1[7]=f2bf(y1.w);
        aq1[8]=f2bf(y2.x); aq1[9]=f2bf(y2.y); aq1[10]=f2bf(y2.z); aq1[11]=f2bf(y2.w);
        aq1[12]=f2bf(y3.x); aq1[13]=f2bf(y3.y); aq1[14]=f2bf(y3.z); aq1[15]=f2bf(y3.w);
    }

    v8f oacc[4] = {};
    float mrow[8], lrow[8];
#pragma unroll
    for (int i = 0; i < 8; ++i) { mrow[i] = -1e30f; lrow[i] = 0.f; }

    const int kend = blockIdx.x * 128 + 128;
    const int kk = tid >> 3;                         // 0..31: key row of chunk
    const int db = (tid & 7) * 8;                    // 0..56: d base

    for (int kb = 0; kb < kend; kb += 32) {
        __syncthreads();                             // previous LDS reads done
        {   // cooperative staging: K -> Klds[k][d], V -> Vt[d][k]
            const float* kp = qkv + (size_t)(b * SLEN + kb + kk) * QKVD + NH * HD + kh * HD + db;
            const float* vp = kp + NKV * HD;
            float4 k0 = *(const float4*)kp, k1 = *(const float4*)(kp + 4);
            float4 v0 = *(const float4*)vp, v1 = *(const float4*)(vp + 4);
            float ka[8] = {k0.x, k0.y, k0.z, k0.w, k1.x, k1.y, k1.z, k1.w};
            float va[8] = {v0.x, v0.y, v0.z, v0.w, v1.x, v1.y, v1.z, v1.w};
#pragma unroll
            for (int j = 0; j < 8; ++j) {
                Klds[kk][db + j] = f2bf(ka[j]);
                Vt[db + j][kk]   = f2bf(va[j]);
            }
        }
        __syncthreads();                             // chunk staged
        if (kb > qbase + 15) continue;               // causal: wave done (still barriers)

        // ---- scores: S = Q @ K^T  (two 16x16 tiles, 2 wmma each over d halves) ----
        v8f sacc[2] = {};
#pragma unroll
        for (int kt = 0; kt < 2; ++kt) {
            const bf16* kpl = &Klds[kt * 16 + l16][0];
#pragma unroll
            for (int dh = 0; dh < 2; ++dh) {
                const bf16* p = kpl + dh * 32 + grp;
                v8bf lo = *(const v8bf*)p;
                v8bf hi = *(const v8bf*)(p + 16);
                v16bf bf;
#pragma unroll
                for (int j = 0; j < 8; ++j) { bf[j] = lo[j]; bf[8 + j] = hi[j]; }
                sacc[kt] = __builtin_amdgcn_wmma_f32_16x16x32_bf16(
                    false, dh ? aq1 : aq0, false, bf, (short)0, sacc[kt], false, false);
            }
        }

        // ---- causal mask + online softmax (row stats across 16-lane groups) ----
        float s0[8], s1[8];
#pragma unroll
        for (int i = 0; i < 8; ++i) {
            int q = qbase + rh + i;
            int kc = kb + l16;
            s0[i] = (kc      <= q) ? sacc[0][i] * 0.125f : -1e30f;
            s1[i] = (kc + 16 <= q) ? sacc[1][i] * 0.125f : -1e30f;
        }
#pragma unroll
        for (int i = 0; i < 8; ++i) {
            float mloc = fmaxf(s0[i], s1[i]);
#pragma unroll
            for (int off = 8; off >= 1; off >>= 1)
                mloc = fmaxf(mloc, __shfl_xor(mloc, off, 32));
            float mnew  = fmaxf(mrow[i], mloc);
            float scale = __expf(mrow[i] - mnew);
            float p0 = __expf(s0[i] - mnew);
            float p1 = __expf(s1[i] - mnew);
            float rs = p0 + p1;
#pragma unroll
            for (int off = 8; off >= 1; off >>= 1)
                rs += __shfl_xor(rs, off, 32);
            lrow[i] = lrow[i] * scale + rs;
            mrow[i] = mnew;
#pragma unroll
            for (int t = 0; t < 4; ++t) oacc[t][i] *= scale;
            Plds[wave][rh + i][l16]      = f2bf(p0);
            Plds[wave][rh + i][16 + l16] = f2bf(p1);
        }

        // ---- O += P @ V (re-stripe P via LDS; per-wave in-order DS) ----
        v16bf pf;
        {
            const bf16* pp = &Plds[wave][l16][grp];
            v8bf lo = *(const v8bf*)pp;
            v8bf hi = *(const v8bf*)(pp + 16);
#pragma unroll
            for (int j = 0; j < 8; ++j) { pf[j] = lo[j]; pf[8 + j] = hi[j]; }
        }
#pragma unroll
        for (int t = 0; t < 4; ++t) {
            const bf16* vpl = &Vt[t * 16 + l16][grp];
            v8bf lo = *(const v8bf*)vpl;
            v8bf hi = *(const v8bf*)(vpl + 16);
            v16bf vf;
#pragma unroll
            for (int j = 0; j < 8; ++j) { vf[j] = lo[j]; vf[8 + j] = hi[j]; }
            oacc[t] = __builtin_amdgcn_wmma_f32_16x16x32_bf16(
                false, pf, false, vf, (short)0, oacc[t], false, false);
        }
    }

    // ---- write normalized output ----
#pragma unroll
    for (int t = 0; t < 4; ++t) {
#pragma unroll
        for (int i = 0; i < 8; ++i) {
            int q = qbase + rh + i;
            attn[(size_t)(b * SLEN + q) * (NH * HD) + head * HD + t * 16 + l16]
                = oacc[t][i] / lrow[i];
        }
    }
}

// -------------------- driver --------------------
extern "C" void kernel_launch(void* const* d_in, const int* in_sizes, int n_in,
                              void* d_out, int out_size, void* d_ws, size_t ws_size,
                              hipStream_t stream) {
    const int*   ids    = (const int*)d_in[0];
    const int*   pos    = (const int*)d_in[1];
    const float* emb    = (const float*)d_in[2];
    const float* w_qkv  = (const float*)d_in[3];
    const float* w_o    = (const float*)d_in[4];
    const float* w_gu   = (const float*)d_in[5];
    const float* w_dn   = (const float*)d_in[6];
    const float* ln1    = (const float*)d_in[7];
    const float* ln2    = (const float*)d_in[8];
    const float* normw  = (const float*)d_in[9];
    float* out = (float*)d_out;
    float* ws  = (float*)d_ws;

    // workspace layout (float elements); act aliases the dead qkv+attn region
    float* resid = ws;                       // 4,194,304
    float* hn    = ws + 4194304;             // 4,194,304
    float* qkv   = ws + 8388608;             // 5,242,880
    float* attn  = ws + 13631488;            // 4,194,304
    float* act   = ws + 8388608;             // 11,534,336 (aliases qkv+attn, both dead)
    float* hbuf  = ws + 19922944;            // 4,194,304  -> total 96.5 MB

    embed_kernel<<<NTOK, 256, 0, stream>>>(ids, emb, resid);

    for (int l = 0; l < 2; ++l) {
        rmsnorm_kernel<<<NTOK, 256, 0, stream>>>(resid, l == 0 ? nullptr : hbuf,
                                                 ln1 + (size_t)l * Hdim, hn);
        gemm_kernel<0><<<dim3(QKVD / 64, NTOK / 128), 256, 0, stream>>>(
            hn, w_qkv + (size_t)l * Hdim * QKVD, qkv, NTOK, QKVD, Hdim);
        rope_kernel<<<(NTOK * (NH + NKV) * 32) / 256, 256, 0, stream>>>(qkv, pos);
        attn_kernel<<<dim3(SLEN / 128, Bsz * NH), 256, 0, stream>>>(qkv, attn);
        gemm_kernel<1><<<dim3(Hdim / 64, NTOK / 128), 256, 0, stream>>>(
            attn, w_o + (size_t)l * Hdim * Hdim, resid, NTOK, Hdim, Hdim);
        rmsnorm_kernel<<<NTOK, 256, 0, stream>>>(resid, nullptr,
                                                 ln2 + (size_t)l * Hdim, hn);
        gemm_gateup_kernel<<<dim3(Idim / 64, NTOK / 128), 256, 0, stream>>>(
            hn, w_gu + (size_t)l * Hdim * 2 * Idim, act);
        gemm_kernel<0><<<dim3(Hdim / 64, NTOK / 128), 256, 0, stream>>>(
            act, w_dn + (size_t)l * Idim * Hdim, hbuf, NTOK, Hdim, Idim);
    }

    rmsnorm_kernel<<<NTOK, 256, 0, stream>>>(resid, hbuf, normw, out);
    (void)in_sizes; (void)n_in; (void)out_size; (void)ws_size;
}
